// Attention_49684181680760
// MI455X (gfx1250) — compile-verified, matrix operand order checked by reading, not run on hardware
//
#include <hip/hip_runtime.h>

// ---------------------------------------------------------------------------
// Types for CDNA5 WMMA (wave32): bf16 16x16x32, f32 accumulate
// ---------------------------------------------------------------------------
typedef __attribute__((ext_vector_type(16))) __bf16 v16bf;
typedef __attribute__((ext_vector_type(8)))  float  v8f;

union AFrag {
    v16bf v;
    uint4 u[2];
    unsigned short s[16];
};

__device__ __forceinline__ unsigned short f32_to_bf16(float f) {
    unsigned int u = __float_as_uint(f);
    unsigned int rounding = 0x7FFFu + ((u >> 16) & 1u);
    u += rounding;
    return (unsigned short)(u >> 16);
}

__device__ __forceinline__ v8f wmma_bf16(const AFrag& a, const AFrag& b, v8f c) {
    return __builtin_amdgcn_wmma_f32_16x16x32_bf16(
        /*neg_a=*/false, a.v, /*neg_b=*/false, b.v,
        /*c_mod=*/(short)0, c, /*reuse_a=*/false, /*reuse_b=*/false);
}

// Async global->LDS copy (CDNA5, tracked with ASYNCcnt). dst = LDS byte
// address (low 32 bits of a generic pointer to __shared__), src = global ptr.
__device__ __forceinline__ void async_copy_b128(unsigned int lds_addr,
                                                const void* gaddr) {
    asm volatile("global_load_async_to_lds_b128 %0, %1, off"
                 :: "v"(lds_addr), "v"(gaddr) : "memory");
}
__device__ __forceinline__ void wait_asynccnt0() {
    asm volatile("s_wait_asynccnt 0" ::: "memory");
}

// ---------------------------------------------------------------------------
// Elementwise conversions / transposes
// ---------------------------------------------------------------------------
__global__ void convert_f32_bf16(const float* __restrict__ in,
                                 unsigned short* __restrict__ out, int n) {
    int i = blockIdx.x * 256 + threadIdx.x;
    if (i < n) out[i] = f32_to_bf16(in[i]);
}

// W: (K x N) f32 row-major  ->  Wt: (N x K) bf16 row-major
__global__ void w_transpose_bf16(const float* __restrict__ W,
                                 unsigned short* __restrict__ Wt, int K, int N) {
    int i = blockIdx.x * 256 + threadIdx.x;
    if (i >= K * N) return;
    int n = i % N;
    int k = i / N;
    Wt[(size_t)n * K + k] = f32_to_bf16(W[i]);
}

// vraw: (B,T,4,128) f32  ->  vt: (B,4,128,T) bf16
__global__ void v_transpose_bf16(const float* __restrict__ vraw,
                                 unsigned short* __restrict__ vt, int B, int T) {
    int i = blockIdx.x * 256 + threadIdx.x;
    if (i >= B * T * 512) return;
    int c   = i % 512;
    int row = i / 512;
    int t = row % T;
    int b = row / T;
    int kh = c >> 7;
    int h  = c & 127;
    vt[(((size_t)b * 4 + kh) * 128 + h) * T + t] = f32_to_bf16(vraw[i]);
}

// ---------------------------------------------------------------------------
// RMS-norm + RoPE + scale, f32 -> bf16 (one block per token row)
// ---------------------------------------------------------------------------
__global__ void rms_rope_bf16(const float* __restrict__ raw,
                              unsigned short* __restrict__ outb,
                              int rowlen, float scale, int T) {
    int row = blockIdx.x;           // b*T + t
    int t   = row % T;
    const float* r = raw + (size_t)row * rowlen;

    __shared__ float red[8];
    float ss = 0.0f;
    for (int c = threadIdx.x; c < rowlen; c += 256) {
        float v = r[c];
        ss += v * v;
    }
    for (int d = 1; d < 32; d <<= 1) ss += __shfl_xor(ss, d, 32);
    if ((threadIdx.x & 31) == 0) red[threadIdx.x >> 5] = ss;
    __syncthreads();
    float tot = 0.0f;
    for (int i = 0; i < 8; i++) tot += red[i];
    float rms = rsqrtf(tot / (float)rowlen + 1e-6f);

    const float LOG1E4 = 9.210340371976184f;   // ln(10000)
    for (int c = threadIdx.x; c < rowlen; c += 256) {
        int h = c & 127;
        int j = h & 63;
        float angle = (float)t * __expf(-LOG1E4 * (float)j * (1.0f / 64.0f));
        float s, cs;
        __sincosf(angle, &s, &cs);
        float v   = r[c] * rms;
        float rot = (h < 64) ? -r[c + 64] * rms : r[c - 64] * rms;
        outb[(size_t)row * rowlen + c] = f32_to_bf16((v * cs + rot * s) * scale);
    }
}

// ---------------------------------------------------------------------------
// Generic bf16 WMMA GEMM: C[M,N](f32) = A[M,K](bf16,row) @ Bt[N,K](bf16,row)^T
// One wave computes a 32x32 output tile (2x2 wmma fragments).
// ---------------------------------------------------------------------------
__global__ void gemm_bf16_wmma(const unsigned short* __restrict__ A,
                               const unsigned short* __restrict__ Bt,
                               float* __restrict__ C, int M, int N, int K) {
    int wave = blockIdx.x * (blockDim.x >> 5) + (threadIdx.x >> 5);
    int lane = threadIdx.x & 31;
    int lq = lane & 15;
    int hi = lane >> 4;

    int tilesN = N >> 5;
    int tm = (wave / tilesN) << 5;
    int tn = (wave % tilesN) << 5;
    if (tm >= M) return;

    v8f acc[2][2];
    for (int i = 0; i < 2; i++)
        for (int j = 0; j < 2; j++)
            for (int r = 0; r < 8; r++) acc[i][j][r] = 0.0f;

    const unsigned short* a0r = A  + (size_t)(tm + lq)      * K;
    const unsigned short* a1r = A  + (size_t)(tm + 16 + lq) * K;
    const unsigned short* b0r = Bt + (size_t)(tn + lq)      * K;
    const unsigned short* b1r = Bt + (size_t)(tn + 16 + lq) * K;

    for (int k0 = 0; k0 < K; k0 += 32) {
        int base = k0 + hi * 8;
        AFrag a0, a1, b0, b1;
        a0.u[0] = *(const uint4*)(a0r + base); a0.u[1] = *(const uint4*)(a0r + base + 16);
        a1.u[0] = *(const uint4*)(a1r + base); a1.u[1] = *(const uint4*)(a1r + base + 16);
        b0.u[0] = *(const uint4*)(b0r + base); b0.u[1] = *(const uint4*)(b0r + base + 16);
        b1.u[0] = *(const uint4*)(b1r + base); b1.u[1] = *(const uint4*)(b1r + base + 16);
        if (k0 + 32 < K) __builtin_prefetch(a0r + base + 32, 0, 1);

        acc[0][0] = wmma_bf16(a0, b0, acc[0][0]);
        acc[0][1] = wmma_bf16(a0, b1, acc[0][1]);
        acc[1][0] = wmma_bf16(a1, b0, acc[1][0]);
        acc[1][1] = wmma_bf16(a1, b1, acc[1][1]);
    }

    for (int i = 0; i < 2; i++)
        for (int j = 0; j < 2; j++)
            for (int r = 0; r < 8; r++) {
                int row = tm + 16 * i + r + 8 * hi;
                int col = tn + 16 * j + lq;
                C[(size_t)row * N + col] = acc[i][j][r];
            }
}

// ---------------------------------------------------------------------------
// Flash attention with async-LDS K/V staging.
// One block = 8 waves = 8 consecutive query tiles of the SAME (batch, head);
// all waves consume identical K/V chunks, so the block stages each 32-key
// chunk of K (32x128) and V^T (128x32) into LDS once via
// global_load_async_to_lds_b128 (ASYNCcnt), then WMMA fragments are read
// with conflict-free ds_load_b128 (row padding of 8 halfwords).
//
// S^T = K_tile x Q^T so softmax stats are per-lane scalars and the exp'd
// prob tiles are already in PV B-fragment layout (no transposes).
// ---------------------------------------------------------------------------
#define KPAD 136   // 128 + 8 halfwords  (272 B row stride -> 4-bank rotation)
#define VPAD 40    // 32 + 8 halfwords   (80 B row stride  -> 20-bank rotation)

__global__ void attn_wmma(const unsigned short* __restrict__ qb,  // (B,T,2048) bf16 (rope+scale)
                          const unsigned short* __restrict__ kb,  // (B,T,4,128) bf16 (rope)
                          const unsigned short* __restrict__ vt,  // (B,4,128,T) bf16
                          unsigned short* __restrict__ enc) {     // (B,T,2048) bf16
    const int T = 1024, NH = 16, G = 4, KV = 4;
    __shared__ unsigned short kbuf[32 * KPAD];    // 8704 B
    __shared__ unsigned short vbuf[128 * VPAD];   // 10240 B

    int tid  = threadIdx.x;
    int wave = blockIdx.x * 8 + (tid >> 5);
    int lane = tid & 31;
    int lq = lane & 15;
    int hi = lane >> 4;

    int qtile = wave % (T / 16);
    int head  = (wave / (T / 16)) % NH;           // uniform within block
    int b     = wave / ((T / 16) * NH);           // uniform within block
    int kh    = head / G;
    int t0    = qtile * 16;

    // Resident Q^T B-fragments (4 h-slices of 32)
    AFrag qf[4];
    const unsigned short* qrow = qb + ((size_t)(b * T + t0 + lq)) * 2048 + head * 128;
    for (int hs = 0; hs < 4; hs++) {
        int base = hs * 32 + hi * 8;
        qf[hs].u[0] = *(const uint4*)(qrow + base);
        qf[hs].u[1] = *(const uint4*)(qrow + base + 16);
    }

    v8f O[8];
    for (int ht = 0; ht < 8; ht++)
        for (int r = 0; r < 8; r++) O[ht][r] = 0.0f;
    float m = -1e30f, l = 0.0f;

    const unsigned short* kbase = kb + ((size_t)(b * T)) * 512 + kh * 128;
    const unsigned short* vbase = vt + ((size_t)((b * KV + kh) * 128)) * 1024;

    for (int kc = 0; kc < T; kc += 32) {
        // ---- cooperative async staging: K chunk 32x128, V chunk 128x32 ----
        for (int r = 0; r < 2; r++) {             // K: 32 rows x 16 segs of 8
            int task = tid + r * 256;
            int row  = task >> 4;
            int seg  = task & 15;
            async_copy_b128((unsigned int)(size_t)&kbuf[row * KPAD + seg * 8],
                            kbase + (size_t)(kc + row) * 512 + seg * 8);
        }
        for (int r = 0; r < 2; r++) {             // V: 128 rows x 4 segs of 8
            int task = tid + r * 256;
            int row  = task >> 2;
            int seg  = task & 3;
            async_copy_b128((unsigned int)(size_t)&vbuf[row * VPAD + seg * 8],
                            vbase + (size_t)row * 1024 + kc + seg * 8);
        }
        wait_asynccnt0();
        __syncthreads();

        // ---- S^T tiles from LDS: s0 = keys [kc..+15], s1 = keys [kc+16..+31]
        v8f s0, s1;
        for (int r = 0; r < 8; r++) { s0[r] = 0.0f; s1[r] = 0.0f; }
        for (int hs = 0; hs < 4; hs++) {
            int hb = hs * 32 + hi * 8;
            const unsigned short* r0 = &kbuf[lq * KPAD + hb];
            const unsigned short* r1 = &kbuf[(16 + lq) * KPAD + hb];
            AFrag k0f, k1f;
            k0f.u[0] = *(const uint4*)(r0); k0f.u[1] = *(const uint4*)(r0 + 16);
            k1f.u[0] = *(const uint4*)(r1); k1f.u[1] = *(const uint4*)(r1 + 16);
            s0 = wmma_bf16(k0f, qf[hs], s0);
            s1 = wmma_bf16(k1f, qf[hs], s1);
        }

        // ---- online softmax: per-lane = per-query stats (xor-16 combine) ----
        float cm = -1e30f;
        for (int r = 0; r < 8; r++) cm = fmaxf(cm, fmaxf(s0[r], s1[r]));
        cm = fmaxf(cm, __shfl_xor(cm, 16, 32));
        float nm = fmaxf(m, cm);
        float alpha = __expf(m - nm);

        AFrag pf;                      // P^T B-fragment for the PV wmma
        float csum = 0.0f;
        for (int r = 0; r < 8; r++) {
            float p0 = __expf(s0[r] - nm);
            float p1 = __expf(s1[r] - nm);
            csum += p0 + p1;
            pf.s[r]     = f32_to_bf16(p0);
            pf.s[8 + r] = f32_to_bf16(p1);
        }
        csum += __shfl_xor(csum, 16, 32);
        l = l * alpha + csum;
        m = nm;

        // ---- O = O*alpha + V^T_tile x P^T  (O tiles: lane = query, elem = h)
        for (int ht = 0; ht < 8; ht++) {
            const unsigned short* vr = &vbuf[(ht * 16 + lq) * VPAD + hi * 8];
            AFrag vf;
            vf.u[0] = *(const uint4*)(vr);
            vf.u[1] = *(const uint4*)(vr + 16);
            for (int r = 0; r < 8; r++) O[ht][r] *= alpha;
            O[ht] = wmma_bf16(vf, pf, O[ht]);
        }
        __syncthreads();               // LDS reusable for next chunk
    }

    float linv = 1.0f / l;
    unsigned short* erow = enc + ((size_t)(b * T + t0 + lq)) * 2048 + head * 128;
    for (int ht = 0; ht < 8; ht++) {
        union { uint4 q; unsigned short s[8]; } tmp;
        for (int r = 0; r < 8; r++) tmp.s[r] = f32_to_bf16(O[ht][r] * linv);
        *(uint4*)(erow + ht * 16 + hi * 8) = tmp.q;
    }
}

// ---------------------------------------------------------------------------
// Launch
// ---------------------------------------------------------------------------
extern "C" void kernel_launch(void* const* d_in, const int* in_sizes, int n_in,
                              void* d_out, int out_size, void* d_ws, size_t ws_size,
                              hipStream_t stream) {
    const int B = 2, T = 1024, C = 2048;
    const int M = B * T;            // 2048 token rows
    const float qscale = 0.08838834764831845f;  // 1/sqrt(128)

    const float* x   = (const float*)d_in[0];
    const float* wq  = (const float*)d_in[1];   // (C, 2048)
    const float* wk  = (const float*)d_in[2];   // (C, 512)
    const float* wv  = (const float*)d_in[3];   // (C, 512)
    const float* wo  = (const float*)d_in[4];   // (2048, C)
    float* out = (float*)d_out;

    size_t off = 0;
    auto alloc = [&](size_t bytes) -> char* {
        char* p = (char*)d_ws + off;
        off += (bytes + 255) & ~(size_t)255;
        return p;
    };
    unsigned short* x_b   = (unsigned short*)alloc((size_t)M * C * 2);
    unsigned short* wq_t  = (unsigned short*)alloc((size_t)2048 * C * 2);
    unsigned short* wk_t  = (unsigned short*)alloc((size_t)512 * C * 2);
    unsigned short* wv_t  = (unsigned short*)alloc((size_t)512 * C * 2);
    unsigned short* wo_t  = (unsigned short*)alloc((size_t)C * 2048 * 2);
    float* q_raw          = (float*)alloc((size_t)M * 2048 * 4);
    float* k_raw          = (float*)alloc((size_t)M * 512 * 4);
    float* v_raw          = (float*)alloc((size_t)M * 512 * 4);
    unsigned short* q_b   = (unsigned short*)alloc((size_t)M * 2048 * 2);
    unsigned short* k_b   = (unsigned short*)alloc((size_t)M * 512 * 2);
    unsigned short* v_t   = (unsigned short*)alloc((size_t)M * 512 * 2);
    unsigned short* enc   = (unsigned short*)q_raw;   // alias: q_raw dead after postprocess

    // 1) precision conversion + weight transposes
    convert_f32_bf16<<<(M * C + 255) / 256, 256, 0, stream>>>(x, x_b, M * C);
    w_transpose_bf16<<<(C * 2048 + 255) / 256, 256, 0, stream>>>(wq, wq_t, C, 2048);
    w_transpose_bf16<<<(C * 512 + 255) / 256, 256, 0, stream>>>(wk, wk_t, C, 512);
    w_transpose_bf16<<<(C * 512 + 255) / 256, 256, 0, stream>>>(wv, wv_t, C, 512);
    w_transpose_bf16<<<(2048 * C + 255) / 256, 256, 0, stream>>>(wo, wo_t, 2048, C);

    // 2) QKV projections (WMMA)
    gemm_bf16_wmma<<<(M / 32) * (2048 / 32) / 8, 256, 0, stream>>>(x_b, wq_t, q_raw, M, 2048, C);
    gemm_bf16_wmma<<<(M / 32) * (512 / 32) / 8, 256, 0, stream>>>(x_b, wk_t, k_raw, M, 512, C);
    gemm_bf16_wmma<<<(M / 32) * (512 / 32) / 8, 256, 0, stream>>>(x_b, wv_t, v_raw, M, 512, C);

    // 3) RMS-norm + RoPE (+ q scale), V transpose
    rms_rope_bf16<<<M, 256, 0, stream>>>(q_raw, q_b, 2048, qscale, T);
    rms_rope_bf16<<<M, 256, 0, stream>>>(k_raw, k_b, 512, 1.0f, T);
    v_transpose_bf16<<<(M * 512 + 255) / 256, 256, 0, stream>>>(v_raw, v_t, B, T);

    // 4) flash attention (WMMA + async LDS staging): 2048 waves / 8 per block
    attn_wmma<<<(B * 16 * (T / 16)) / 8, 256, 0, stream>>>(q_b, k_b, v_t, enc);

    // 5) output projection (WMMA) -> f32 out
    gemm_bf16_wmma<<<(M / 32) * (C / 32) / 8, 256, 0, stream>>>(enc, wo_t, out, M, C, C);
}
